// BioNet_66039417143416
// MI455X (gfx1250) — matrix operand order repaired and checked by reading, not compiled, not run on hardware
//
#include <hip/hip_runtime.h>

// BioNet steady-state recurrence on MI455X (gfx1250, wave32).
// Sparse block-row WMMA formulation with fully preprocessed operands:
// A-tiles prebuilt in WMMA lane layout (bf16), X carried in bf16 between
// steps, so the 50x-replayed hot loop is branchless loads + v_wmma, and
// the epilogue is a branchless min/cndmask activation (no IEEE div).

#define NN      8192      // nodes
#define B       32        // batch
#define STEPS   50
#define LEAKC   0.01f
#define NBR     512       // block-rows of 16 target rows
#define CAP     512       // edge capacity per block-row (avg 256, +16 sigma)
#define MAXG    (CAP/32)  // edge groups of 32 per block-row

typedef __attribute__((ext_vector_type(16))) __bf16 v16bf;
typedef __attribute__((ext_vector_type(8)))  float  v8f;

__device__ __forceinline__ unsigned short f32_bf16(float f) {
    unsigned int u = __float_as_uint(f);
    u += 0x7FFFu + ((u >> 16) & 1u);          // round-to-nearest-even
    return (unsigned short)(u >> 16);
}

// Branchless Michaelis-Menten-like activation.
// For x>=0: mml(x) = min(x, 1 - 0.25/max(x,0.5))  since x-(1-0.25/x)=(x-.5)^2/x>=0
// and at x<0.5 the sat term evaluates to exactly 0.5 so min() picks x.
__device__ __forceinline__ float mml(float x) {
    float r   = __builtin_amdgcn_rcpf(fmaxf(x, 0.5f));   // v_rcp_f32
    float sat = __builtin_fmaf(-0.25f, r, 1.0f);
    float pos = fminf(x, sat);
    return (x < 0.0f) ? LEAKC * x : pos;                 // v_cndmask
}

// ---- 1. X_bias = X_full^T + bias ; zero bf16 X state --------------------
__global__ void k_init(const float* __restrict__ X_full,
                       const float* __restrict__ bias,
                       float* __restrict__ Xb, unsigned short* __restrict__ Xh0) {
    int tid = blockIdx.x * blockDim.x + threadIdx.x;   // tid = n*32 + b
    int n = tid >> 5, b = tid & 31;
    Xb[tid]  = X_full[b * NN + n] + bias[n];
    Xh0[tid] = 0;                                      // bf16 +0.0
}

// ---- 2. scan mask -> block-row edge records (deterministic, 1 wave/br) --
__global__ void __launch_bounds__(32)
k_build(const float* __restrict__ W, const unsigned char* __restrict__ mask,
        int2* __restrict__ brE, int* __restrict__ brCnt) {
    int br = blockIdx.x, lane = threadIdx.x;
    int cnt = 0;
    for (int r = 0; r < 16; ++r) {
        int row = br * 16 + r;
        const unsigned int* mrow =
            (const unsigned int*)(mask + (size_t)row * NN);   // 8192B aligned
        for (int it = 0; it < NN / 128; ++it) {               // 64 iters/row
            unsigned int m4 = mrow[it * 32 + lane];
            #pragma unroll
            for (int j = 0; j < 4; ++j) {
                bool has = ((m4 >> (8 * j)) & 0xFF) != 0;
                unsigned int bal = __builtin_amdgcn_ballot_w32(has);
                if (has) {
                    int idx = cnt + __popc(bal & ((1u << lane) - 1u));
                    if (idx < CAP) {
                        int col = it * 128 + lane * 4 + j;
                        float wv = W[(size_t)row * NN + col];
                        int2 rec;
                        rec.x = (r << 16) | col;              // tgt_local|src
                        rec.y = __float_as_int(wv);
                        brE[br * CAP + idx] = rec;
                    }
                }
                cnt += __popc(bal);
                if (cnt > CAP) cnt = CAP;
            }
        }
    }
    int padded = (cnt + 31) & ~31;                            // zero-pad tail
    for (int idx = cnt + lane; idx < padded; idx += 32) {
        int2 z; z.x = 0; z.y = 0;
        brE[br * CAP + idx] = z;
    }
    if (lane == 0) brCnt[br] = cnt;
}

// ---- 3. expand records into WMMA-layout A-tiles + src lists (once) ------
// A[m,k] = w_k if tgt_local(k)==m else 0, stored per (lane, element) so the
// hot loop reads it with two coalesced b128 loads. LDS staging keeps all
// loads unconditional (no predicated-load blowup).
__global__ void __launch_bounds__(32)
k_expand(const int2* __restrict__ brE, const int* __restrict__ brCnt,
         unsigned short* __restrict__ Atiles, int* __restrict__ srcs) {
    __shared__ int2 rec[32];
    int br = blockIdx.x, lane = threadIdx.x;
    int half = lane >> 4, m = lane & 15;
    int cnt = brCnt[br];
    int ngroups = (cnt + 31) >> 5;
    for (int g = 0; g < ngroups; ++g) {
        rec[lane] = brE[br * CAP + g * 32 + lane];
        __syncthreads();
        srcs[(br * MAXG + g) * 32 + lane] = rec[lane].x & 0xFFFF;
        union { unsigned short u[16]; uint4 q[2]; } a;
        #pragma unroll
        for (int e = 0; e < 16; ++e) {
            // 16-bit A 16x32 layout (contiguous, matches B): K = e + 16*half
            int kA = e + 16 * half;
            int2 r = rec[kA];                          // unconditional ds load
            int tl = r.x >> 16;
            unsigned short w = f32_bf16(__int_as_float(r.y));
            a.u[e] = (tl == m) ? w : (unsigned short)0;
        }
        uint4* out = (uint4*)(Atiles + ((size_t)(br * MAXG + g) * 32 + lane) * 16);
        out[0] = a.q[0];
        out[1] = a.q[1];
        __syncthreads();
    }
}

// ---- 4. one recurrence step (hot, replayed 50x) -------------------------
// Block = 64 threads = 2 waves; wave w owns batch columns [16w, 16w+16).
// Per group: 2 b128 A loads + 4 b128 src loads + 16 u16 gathers + 1 WMMA.
__global__ void __launch_bounds__(64)
k_step(const unsigned short* __restrict__ Xin,      // bf16 state (n,32)
       float* __restrict__ XoutF,                   // f32 state (final only)
       unsigned short* __restrict__ XoutH,          // bf16 state out
       const float* __restrict__ Xb,
       const unsigned short* __restrict__ Atiles,
       const int* __restrict__ srcs,
       const int* __restrict__ brCnt, int writeF32) {
    int br   = blockIdx.x;
    int w    = threadIdx.x >> 5;                    // wave id: batch half
    int lane = threadIdx.x & 31;
    int half = lane >> 4, m = lane & 15;
    int colb = 16 * w;
    int cnt = brCnt[br];
    int ngroups = (cnt + 31) >> 5;

    v8f acc = {};
    for (int g = 0; g < ngroups; ++g) {
        size_t gb = (size_t)(br * MAXG + g) * 32;
        // A operand: prebuilt, 32B contiguous per lane
        const uint4* ap = (const uint4*)(Atiles + (gb + lane) * 16);
        union { uint4 q[2]; v16bf v; } A;
        A.q[0] = ap[0];
        A.q[1] = ap[1];
        // B operand: lanes 0-15 hold K=0..15, lanes 16-31 hold K=16..31
        const int4* sp4 = (const int4*)(srcs + gb + 16 * half);
        int4 sv[4];
        #pragma unroll
        for (int i = 0; i < 4; ++i) sv[i] = sp4[i];
        const int* sl = (const int*)sv;
        union { unsigned short u[16]; v16bf v; } Bop;
        #pragma unroll
        for (int e = 0; e < 16; ++e)
            Bop.u[e] = Xin[sl[e] * B + colb + m];   // bf16 gather, no convert
        acc = __builtin_amdgcn_wmma_f32_16x16x32_bf16(
                  false, A.v, false, Bop.v, (short)0, acc, false, false);
    }

    // epilogue: D layout (32-bit C/D 16x16): row = v + 8*half, col = m
    int rbase = br * 16 + 8 * half;
    int c     = colb + m;
    float y[8];
    #pragma unroll
    for (int v = 0; v < 8; ++v) {
        float x = acc[v] + Xb[(rbase + v) * B + c];
        y[v] = mml(x);                              // branchless
    }
    #pragma unroll
    for (int v = 0; v < 8; ++v)
        XoutH[(rbase + v) * B + c] = f32_bf16(y[v]);
    if (writeF32 != 0) {                            // uniform, hoisted
        #pragma unroll
        for (int v = 0; v < 8; ++v)
            XoutF[(rbase + v) * B + c] = y[v];
    }
}

// ---- 5. output transpose: (n,B) -> (B,n) --------------------------------
__global__ void k_out(const float* __restrict__ X, float* __restrict__ out) {
    int tid = blockIdx.x * blockDim.x + threadIdx.x;   // tid = b*8192 + n
    int b = tid >> 13, n = tid & 8191;
    out[tid] = X[n * B + b];                           // coalesced writes
}

extern "C" void kernel_launch(void* const* d_in, const int* in_sizes, int n_in,
                              void* d_out, int out_size, void* d_ws, size_t ws_size,
                              hipStream_t stream) {
    const float*         X_full = (const float*)d_in[0];         // (32, 8192)
    const float*         W      = (const float*)d_in[1];         // (8192, 8192)
    const float*         bias   = (const float*)d_in[2];         // (8192, 1)
    const unsigned char* mask   = (const unsigned char*)d_in[3]; // bool

    char* ws = (char*)d_ws;                              // ~16 MB total
    float*          Xb     = (float*)(ws);               // 1 MB @ 0
    float*          Xf0    = (float*)(ws + (1 << 20));   // 1 MB
    float*          Xf1    = (float*)(ws + (2 << 20));   // 1 MB
    unsigned short* Xh0    = (unsigned short*)(ws + (3 << 20));              // 512 KB
    unsigned short* Xh1    = (unsigned short*)(ws + (3 << 20) + (1 << 19));  // 512 KB
    int*            brCnt  = (int*)(ws + (4 << 20));                         // 2 KB
    int2*           brE    = (int2*)(ws + (4 << 20) + (1 << 16));            // 2 MB
    int*            srcs   = (int*)(ws + (7 << 20));                         // 1 MB
    unsigned short* Atiles = (unsigned short*)(ws + (8 << 20));              // 8 MB

    k_init<<<(NN * B) / 256, 256, 0, stream>>>(X_full, bias, Xb, Xh0);
    k_build<<<NBR, 32, 0, stream>>>(W, mask, brE, brCnt);
    k_expand<<<NBR, 32, 0, stream>>>(brE, brCnt, Atiles, srcs);

    float*          Xf[2] = { Xf0, Xf1 };
    unsigned short* Xh[2] = { Xh0, Xh1 };
    int cur = 0;
    for (int t = 0; t < STEPS; ++t) {
        int nxt = cur ^ 1;
        k_step<<<NBR, 64, 0, stream>>>(Xh[cur], Xf[nxt], Xh[nxt], Xb,
                                       Atiles, srcs, brCnt,
                                       (t == STEPS - 1) ? 1 : 0);
        cur = nxt;
    }
    k_out<<<(NN * B) / 256, 256, 0, stream>>>(Xf[cur], (float*)d_out);
}
